// TreeRouter_54288386622043
// MI455X (gfx1250) — compile-verified, matrix operand order checked by reading, not run on hardware
//
#include <hip/hip_runtime.h>
#include <hip/hip_bf16.h>
#include <stdint.h>

// ---------------------------------------------------------------------------
// TreeRouter fused implementation for gfx1250 (MI455X).
// Dominant cost: einsum('nd,cdh->nch') = 137 GFLOP -> bf16 WMMA
// (v_wmma_f32_16x16x32_bf16), fused with LayerNorm + GELU + tiny GEMM2 so
// the 268MB intermediate never leaves the WGP. 64-row M-tiles per block give
// 4x W1 reuse (L2 traffic ~2.4GB vs ~10GB at 16-row tiles), balancing data
// movement against the ~27us of dense-bf16 matrix compute.
// ---------------------------------------------------------------------------

typedef __attribute__((ext_vector_type(16))) __bf16 v16bf;
typedef __attribute__((ext_vector_type(8)))  float v8f;

#define NTOK   8192      // B*S
#define DDIM   1024
#define HDIM_  512
#define ADIM   256
#define CCL    16
#define EEX    8

static __device__ __forceinline__ uint16_t f2bf(float f) {
    uint32_t u = __float_as_uint(f);
    uint32_t r = u + 0x7FFFu + ((u >> 16) & 1u);   // round-to-nearest-even
    return (uint16_t)(r >> 16);
}

static __device__ __forceinline__ float gelu_exact(float x) {
    return 0.5f * x * (1.0f + erff(x * 0.70710678118654752f));
}

// ---------------- conversion kernels -----------------------------------

__global__ void cvt_flat(const float* __restrict__ in, uint16_t* __restrict__ out, long n) {
    long i = (long)blockIdx.x * blockDim.x + threadIdx.x;
    if (i < n) out[i] = f2bf(in[i]);
}

// in: (batch, R, C) f32 row-major ; out: (batch, C, R) bf16 (transposed)
__global__ void cvt_transpose_bf16(const float* __restrict__ in, uint16_t* __restrict__ out,
                                   int R, int C) {
    long per = (long)R * C;
    in  += (long)blockIdx.z * per;
    out += (long)blockIdx.z * per;
    long i = (long)blockIdx.x * blockDim.x + threadIdx.x;
    if (i >= per) return;
    int c = (int)(i / R);
    int r = (int)(i - (long)c * R);
    out[(long)c * R + r] = f2bf(in[(long)r * C + c]);
}

// in: (batch, R, C) f32 row-major ; out: (batch, C, R) f32 (transposed)
__global__ void cvt_transpose_f32(const float* __restrict__ in, float* __restrict__ out,
                                  int R, int C) {
    long per = (long)R * C;
    in  += (long)blockIdx.z * per;
    out += (long)blockIdx.z * per;
    long i = (long)blockIdx.x * blockDim.x + threadIdx.x;
    if (i >= per) return;
    int c = (int)(i / R);
    int r = (int)(i - (long)c * R);
    out[(long)c * R + r] = in[(long)r * C + c];
}

// ---------------- fused GEMM1 -> LN -> GELU -> GEMM2 --------------------
// Block: 256 threads (8 waves). Each block: MT*16 token rows, full HDIM cols.
// Wave w owns columns [w*NT*16, (w+1)*NT*16). 16 WMMAs per wave per k-step
// at MT=NT=4 -> 4x register reuse of both A and B fragments.
// grid.y = batch (cluster index for the expert path, 1 otherwise).

union BFrag { v16bf v; uint32_t u[8]; };

template<int HDIM, int OUTDIM, int MT, int NT, bool SIGMOID>
__global__ __launch_bounds__(256)
void fused_mlp(const uint16_t* __restrict__ xb,                    // N x D bf16
               const uint16_t* __restrict__ w1t, long w1Stride,    // (batch) HDIM x D bf16 (K-contig)
               const float* __restrict__ b1, const float* __restrict__ g,
               const float* __restrict__ bt, long vecStride,
               const float* __restrict__ w2t, long w2Stride,       // (batch) OUTDIM x HDIM f32
               const float* __restrict__ b2, long b2Stride,
               const float* __restrict__ rel, long relStride,
               const int* __restrict__ tempPtr,
               float* __restrict__ out, long outRowStride, int D) {
    static_assert(8 * NT * 16 == HDIM, "wave tiling must cover HDIM");
    constexpr int HP = HDIM + 4;                  // LDS pad: conflict-free
    __shared__ float hbuf[16 * HP];               // one 16-row chunk at a time

    const int c    = blockIdx.y;
    const int n0   = blockIdx.x * (MT * 16);
    const int lane = threadIdx.x & 31;
    const int wave = threadIdx.x >> 5;
    const int m    = lane & 15;                   // row (A) / col (B) within tile
    const int hiK  = (lane >> 4) * 8;             // K sub-offset for hi half-wave

    const uint16_t* W1  = w1t + (long)c * w1Stride;
    const float*    B1  = b1  + (long)c * vecStride;
    const float*    G   = g   + (long)c * vecStride;
    const float*    BT  = bt  + (long)c * vecStride;
    const float*    W2T = w2t + (long)c * w2Stride;
    const float*    B2  = b2  + (long)c * b2Stride;

    // ---------------- GEMM1: (MT*16) x HDIM = x @ W1^T -------------------
    v8f acc[MT][NT];
#pragma unroll
    for (int mt = 0; mt < MT; ++mt)
#pragma unroll
        for (int t = 0; t < NT; ++t) acc[mt][t] = (v8f){0,0,0,0,0,0,0,0};

    for (int kk = 0; kk < D; kk += 32) {
        BFrag bfr[NT];
#pragma unroll
        for (int t = 0; t < NT; ++t) {            // B 32x16 fragments, K-contig
            const int col = wave * (NT * 16) + t * 16 + m;
            const uint16_t* bp = W1 + (long)col * D + kk + hiK;
            uint4 lo = *(const uint4*)(bp);
            uint4 hi = *(const uint4*)(bp + 16);
            bfr[t].u[0]=lo.x; bfr[t].u[1]=lo.y; bfr[t].u[2]=lo.z; bfr[t].u[3]=lo.w;
            bfr[t].u[4]=hi.x; bfr[t].u[5]=hi.y; bfr[t].u[6]=hi.z; bfr[t].u[7]=hi.w;
        }
#pragma unroll
        for (int mt = 0; mt < MT; ++mt) {         // A 16x32 fragment per row tile
            const uint16_t* ap = xb + (long)(n0 + mt * 16 + m) * D + kk + hiK;
            BFrag a;
            uint4 lo = *(const uint4*)(ap);
            uint4 hi = *(const uint4*)(ap + 16);
            a.u[0]=lo.x; a.u[1]=lo.y; a.u[2]=lo.z; a.u[3]=lo.w;
            a.u[4]=hi.x; a.u[5]=hi.y; a.u[6]=hi.z; a.u[7]=hi.w;
#pragma unroll
            for (int t = 0; t < NT; ++t)
                acc[mt][t] = __builtin_amdgcn_wmma_f32_16x16x32_bf16(
                    false, a.v, false, bfr[t].v, (short)0, acc[mt][t], false, false);
        }
    }

    const float invT  = tempPtr ? (1.0f / (float)(*tempPtr)) : 1.0f;
    const float scale = (rel ? 1.0f : 1.0f);      // rel applied per-output below

    // ---------------- per-16-row-chunk epilogue ---------------------------
#pragma unroll 1
    for (int mt = 0; mt < MT; ++mt) {
        // spill accumulators (+bias) to LDS: D layout VGPR j -> row j + hiK
#pragma unroll
        for (int t = 0; t < NT; ++t) {
            const int col = wave * (NT * 16) + t * 16 + m;
            const float bias = B1[col];
#pragma unroll
            for (int j = 0; j < 8; ++j) {
                const int r = j + hiK;
                hbuf[r * HP + col] = acc[mt][t][j] + bias;
            }
        }
        __syncthreads();

        // LayerNorm + GELU over HDIM per row (16 rows x 16 threads)
        {
            const int r   = threadIdx.x >> 4;
            const int sub = threadIdx.x & 15;
            float s = 0.f, sq = 0.f;
            for (int k = sub; k < HDIM; k += 16) {
                float v = hbuf[r * HP + k];
                s += v; sq += v * v;
            }
#pragma unroll
            for (int msk = 8; msk >= 1; msk >>= 1) {
                s  += __shfl_xor(s,  msk);
                sq += __shfl_xor(sq, msk);
            }
            const float mu   = s * (1.0f / HDIM);
            const float var  = sq * (1.0f / HDIM) - mu * mu;
            const float rsig = rsqrtf(var + 1e-5f);
            for (int k = sub; k < HDIM; k += 16) {
                float v = hbuf[r * HP + k];
                v = (v - mu) * rsig * G[k] + BT[k];
                hbuf[r * HP + k] = gelu_exact(v);
            }
        }
        __syncthreads();

        // GEMM2: (16 x HDIM) @ W2T(OUTDIM x HDIM)^T, float4 streamed
        if (threadIdx.x < 16 * OUTDIM) {
            const int r = threadIdx.x & 15;
            const int o = threadIdx.x >> 4;
            const float* w2row = W2T + (long)o * HDIM;
            float dot = 0.f;
            for (int k = 0; k < HDIM; k += 4) {
                float4 hv = *(const float4*)&hbuf[r * HP + k];
                float4 wv = *(const float4*)&w2row[k];
                dot = fmaf(hv.x, wv.x, dot);
                dot = fmaf(hv.y, wv.y, dot);
                dot = fmaf(hv.z, wv.z, dot);
                dot = fmaf(hv.w, wv.w, dot);
            }
            float val = dot + B2[o];
            if (rel) val *= rel[(long)c * relStride + o];
            val *= invT * scale;
            if (SIGMOID) val = 1.0f / (1.0f + expf(-val));
            out[(long)(n0 + mt * 16 + r) * outRowStride + (long)c * OUTDIM + o] = val;
        }
        __syncthreads();   // hbuf reused by next chunk
    }
}

// ---------------- routing: softmax + top-2 (stable ties like lax.top_k) --

__global__ void routing_kernel(const float* __restrict__ clog,   // N x 16 (already *rel/temp)
                               const float* __restrict__ el,     // N x 16 x 8 (already *rel)
                               const int* __restrict__ tempPtr,
                               float* __restrict__ out, int N) {
    int n = blockIdx.x * blockDim.x + threadIdx.x;
    if (n >= N) return;
    const float invT = 1.0f / (float)(*tempPtr);

    // cluster softmax over 16
    float p[CCL];
    float mx = -3.4e38f;
    for (int j = 0; j < CCL; ++j) { p[j] = clog[(long)n * CCL + j]; mx = fmaxf(mx, p[j]); }
    float sum = 0.f;
    for (int j = 0; j < CCL; ++j) { p[j] = expf(p[j] - mx); sum += p[j]; }
    float inv = 1.0f / sum;
    for (int j = 0; j < CCL; ++j) p[j] *= inv;

    int   bi = 0;       float bv = p[0];
    for (int j = 1; j < CCL; ++j) if (p[j] > bv) { bv = p[j]; bi = j; }
    int   si = (bi == 0) ? 1 : 0; float sv = p[si];
    for (int j = si + 1; j < CCL; ++j) if (j != bi && p[j] > sv) { sv = p[j]; si = j; }

    const long Nl = N;
    out[            (long)n * 2 + 0] = bv;
    out[            (long)n * 2 + 1] = sv;
    out[2 * Nl +    (long)n * 2 + 0] = (float)bi;
    out[2 * Nl +    (long)n * 2 + 1] = (float)si;

    int ci2[2] = { bi, si };
#pragma unroll
    for (int i = 0; i < 2; ++i) {
        const float* base = el + ((long)n * CCL + ci2[i]) * EEX;
        float e[EEX];
        float em = -3.4e38f;
        for (int j = 0; j < EEX; ++j) { e[j] = base[j] * invT; em = fmaxf(em, e[j]); }
        float es = 0.f;
        for (int j = 0; j < EEX; ++j) { e[j] = expf(e[j] - em); es += e[j]; }
        float einv = 1.0f / es;
        for (int j = 0; j < EEX; ++j) e[j] *= einv;

        int   b2i = 0;  float b2v = e[0];
        for (int j = 1; j < EEX; ++j) if (e[j] > b2v) { b2v = e[j]; b2i = j; }
        int   s2i = (b2i == 0) ? 1 : 0; float s2v = e[s2i];
        for (int j = s2i + 1; j < EEX; ++j) if (j != b2i && e[j] > s2v) { s2v = e[j]; s2i = j; }

        out[4 * Nl + (long)i * 2 * Nl + (long)n * 2 + 0] = b2v;
        out[4 * Nl + (long)i * 2 * Nl + (long)n * 2 + 1] = s2v;
        out[8 * Nl + (long)i * 2 * Nl + (long)n * 2 + 0] = (float)b2i;
        out[8 * Nl + (long)i * 2 * Nl + (long)n * 2 + 1] = (float)s2i;
    }
}

// ---------------------------------------------------------------------------

extern "C" void kernel_launch(void* const* d_in, const int* in_sizes, int n_in,
                              void* d_out, int out_size, void* d_ws, size_t ws_size,
                              hipStream_t stream) {
    (void)in_sizes; (void)n_in; (void)out_size; (void)ws_size;

    const float* x           = (const float*)d_in[0];
    const float* cw1         = (const float*)d_in[1];
    const float* cb1         = (const float*)d_in[2];
    const float* cg          = (const float*)d_in[3];
    const float* cbt         = (const float*)d_in[4];
    const float* cw2         = (const float*)d_in[5];
    const float* cb2         = (const float*)d_in[6];
    const float* ew1         = (const float*)d_in[7];
    const float* eb1         = (const float*)d_in[8];
    const float* eg          = (const float*)d_in[9];
    const float* ebt         = (const float*)d_in[10];
    const float* ew2         = (const float*)d_in[11];
    const float* eb2         = (const float*)d_in[12];
    const float* aw1         = (const float*)d_in[13];
    const float* ab1         = (const float*)d_in[14];
    const float* ag          = (const float*)d_in[15];
    const float* abt         = (const float*)d_in[16];
    const float* aw2         = (const float*)d_in[17];
    const float* ab2         = (const float*)d_in[18];
    const float* cluster_rel = (const float*)d_in[19];
    const float* expert_rel  = (const float*)d_in[20];
    const int*   tempPtr     = (const int*)d_in[23];

    float* out = (float*)d_out;
    const int N = NTOK;

    // workspace carve-up (256B aligned)
    uint8_t* ws = (uint8_t*)d_ws;
    size_t off = 0;
    auto carve = [&](size_t bytes) -> uint8_t* {
        uint8_t* p = ws + off;
        off += (bytes + 255) & ~(size_t)255;
        return p;
    };
    uint16_t* xb    = (uint16_t*)carve((size_t)N * DDIM * 2);
    uint16_t* cw1T  = (uint16_t*)carve((size_t)HDIM_ * DDIM * 2);
    uint16_t* ew1T  = (uint16_t*)carve((size_t)CCL * HDIM_ * DDIM * 2);
    uint16_t* aw1T  = (uint16_t*)carve((size_t)ADIM * DDIM * 2);
    float*    cw2T  = (float*)carve((size_t)CCL * HDIM_ * 4);
    float*    ew2T  = (float*)carve((size_t)CCL * EEX * HDIM_ * 4);
    float*    aw2T  = (float*)carve((size_t)ADIM * 4);
    float*    clog  = (float*)carve((size_t)N * CCL * 4);
    float*    elall = (float*)carve((size_t)N * CCL * EEX * 4);

    // 1) bf16 conversions (x row-major; W1 weights transposed -> K contiguous)
    {
        long n = (long)N * DDIM;
        cvt_flat<<<dim3((unsigned)((n + 255) / 256)), 256, 0, stream>>>(x, xb, n);
    }
    cvt_transpose_bf16<<<dim3((DDIM * HDIM_ + 255) / 256, 1, 1),   256, 0, stream>>>(cw1, cw1T, DDIM, HDIM_);
    cvt_transpose_bf16<<<dim3((DDIM * HDIM_ + 255) / 256, 1, CCL), 256, 0, stream>>>(ew1, ew1T, DDIM, HDIM_);
    cvt_transpose_bf16<<<dim3((DDIM * ADIM  + 255) / 256, 1, 1),   256, 0, stream>>>(aw1, aw1T, DDIM, ADIM);
    // W2 transposed f32 (tiny) -> contiguous dot rows for the GEMM2 epilogue
    cvt_transpose_f32<<<dim3((HDIM_ * CCL + 255) / 256, 1, 1),   256, 0, stream>>>(cw2, cw2T, HDIM_, CCL);
    cvt_transpose_f32<<<dim3((HDIM_ * EEX + 255) / 256, 1, CCL), 256, 0, stream>>>(ew2, ew2T, HDIM_, EEX);
    cvt_transpose_f32<<<dim3((ADIM * 1    + 255) / 256, 1, 1),   256, 0, stream>>>(aw2, aw2T, ADIM, 1);

    // 2) cluster path: (N x 1024) @ (1024 x 512) -> LN -> GELU -> @ (512 x 16)
    fused_mlp<HDIM_, CCL, 4, 4, false><<<dim3(N / 64, 1), 256, 0, stream>>>(
        xb, cw1T, 0, cb1, cg, cbt, 0, cw2T, 0, cb2, 0,
        cluster_rel, 0, tempPtr, clog, CCL, DDIM);

    // 3) expert path (dominant): per cluster c, fused GEMM1+LN+GELU+GEMM2
    fused_mlp<HDIM_, EEX, 4, 4, false><<<dim3(N / 64, CCL), 256, 0, stream>>>(
        xb, ew1T, (long)HDIM_ * DDIM, eb1, eg, ebt, HDIM_,
        ew2T, (long)EEX * HDIM_, eb2, EEX,
        expert_rel, EEX, nullptr, elall, (long)CCL * EEX, DDIM);

    // 4) adaptive path -> sigmoid written straight into out[12N .. 13N)
    fused_mlp<ADIM, 1, 4, 2, true><<<dim3(N / 64, 1), 256, 0, stream>>>(
        xb, aw1T, 0, ab1, ag, abt, 0, aw2T, 0, ab2, 0,
        nullptr, 0, nullptr, out + 12 * (long)N, 1, DDIM);

    // 5) routing: softmax + top-2 (clusters then experts), indices as floats
    routing_kernel<<<dim3(N / 256), 256, 0, stream>>>(clog, elall, tempPtr, out, N);
}
